// MultiHeadAttention_12283606467896
// MI455X (gfx1250) — compile-verified, matrix operand order checked by reading, not run on hardware
//
#include <hip/hip_runtime.h>
#include <hip/hip_bf16.h>

typedef __bf16 v16bf __attribute__((ext_vector_type(16)));
typedef float  v8f   __attribute__((ext_vector_type(8)));
typedef unsigned int v4u __attribute__((ext_vector_type(4)));
typedef int          v8i __attribute__((ext_vector_type(8)));
typedef int          v4i __attribute__((ext_vector_type(4)));

#define D_MODEL  2048
#define NUM_HEADS 16
#define D_HEAD   128
#define BATCH    2
#define SEQ      2048
#define BH       (BATCH * NUM_HEADS)

#if __has_builtin(__builtin_amdgcn_tensor_load_to_lds)
#define HAVE_TDM 1
#else
#define HAVE_TDM 0
#endif

union Frag { v16bf bf; unsigned int u[8]; };
union PkBf { __bf16 h[2]; unsigned int u; };

// K-index of the bf16 pair held in VGPR v by this lane (16x32 A/B fragment,
// ISA 7.12.2): lanes 0-15 hold K 0..7 & 16..23, lanes 16-31 hold K 8..15 &
// 24..31, two consecutive K per VGPR -> per-lane contiguous 16B chunks, which
// the compiler merges into global_load_b128 / ds_load_b128.
__device__ __forceinline__ int frag_k(int v, int lane) {
    int kb = (lane & 16) ? 8 : 0;
    return (v < 4) ? (kb + 2 * v) : (16 + kb + 2 * (v - 4));
}

__device__ __forceinline__ Frag load_frag(const __bf16* base, int ld, int lane) {
    Frag f;
    const int r = lane & 15;
#pragma unroll
    for (int v = 0; v < 8; ++v) {
        int k = frag_k(v, lane);
        f.u[v] = *(const unsigned int*)(base + (size_t)r * ld + k);
    }
    return f;
}

__device__ __forceinline__ v8f wmma_bf16(const Frag& a, const Frag& b, v8f c) {
    return __builtin_amdgcn_wmma_f32_16x16x32_bf16(
        false, a.bf, false, b.bf, (short)0, c, false, false);
}

// Issue one TDM 2-D tile load: tile_w x tile_h bf16 elements, row stride
// `stride` elements, global -> LDS byte offset lds_off. D# per ISA 08 §8.
__device__ __forceinline__ void tdm_load_tile_bf16(const __bf16* gsrc,
                                                   unsigned lds_off,
                                                   int tile_w, int tile_h,
                                                   int stride) {
#if HAVE_TDM
    unsigned long long ga = (unsigned long long)(const void*)gsrc;
    v4u g0;
    g0[0] = 0x1u;                                        // count=1, user D#
    g0[1] = lds_off;                                     // lds_addr (bytes)
    g0[2] = (unsigned)(ga & 0xffffffffu);                // global_addr[31:0]
    g0[3] = (unsigned)((ga >> 32) & 0x01ffffffu)         // global_addr[56:32]
          | (2u << 30);                                  // type = 2 (image)
    v8i g1;
    g1[0] = 0x10000;                                     // data_size=1 (2B)
    g1[1] = (tile_w & 0xffff) << 16;                     // tensor_dim0 lo16
    g1[2] = ((unsigned)tile_w >> 16) | ((tile_h & 0xffff) << 16); // d0 hi/d1 lo
    g1[3] = ((unsigned)tile_h >> 16) | ((tile_w & 0xffff) << 16); // d1 hi/tile_dim0
    g1[4] = (tile_h & 0xffff);                           // tile_dim1 (dim2=0)
    g1[5] = stride;                                      // tensor_dim0_stride lo32
    g1[6] = 0;
    g1[7] = 0;
    v4i z4 = {0, 0, 0, 0};
#if __clang_major__ >= 23
    v8i z8 = {0, 0, 0, 0, 0, 0, 0, 0};
    __builtin_amdgcn_tensor_load_to_lds(g0, g1, z4, z4, z8, 0);
#else
    __builtin_amdgcn_tensor_load_to_lds(g0, g1, z4, z4, 0);
#endif
#endif
}

// ---------------------------------------------------------------------------
// Projection GEMM:  C(M x N) = A(M x K) * W(K x N) + bias
//   AMODE: 0 = A fp32 (manual convert-stage), 1 = A bf16 (TDM async tile load)
//   OMODE: 0 = bf16 head-split (BH,S,Dh); 1 = f32 plain; 2 = bf16 (BH,Dh,S)
// Block tile 128(M) x 64(N), 256 threads = 8 waves (4x2), 2x2 WMMA tiles/wave.
// ---------------------------------------------------------------------------
template <int AMODE, int OMODE>
__global__ __launch_bounds__(256)
void proj_gemm(const void* __restrict__ Aptr, const float* __restrict__ W,
               const float* __restrict__ bias, void* __restrict__ Out) {
    __shared__ __align__(16) __bf16 smemP[128 * 32 + 64 * 32];
    __bf16* sA = smemP;              // [m][k]  (LDS byte offset 0)
    __bf16* sB = smemP + 128 * 32;   // [n][k]  transposed

    const int tid  = threadIdx.x;
    const int lane = tid & 31;
    const int w    = tid >> 5;
    const int wm   = w >> 1;
    const int wn   = w & 1;
    const int m0   = blockIdx.y * 128;
    const int n0   = blockIdx.x * 64;

    v8f acc[2][2] = {};

    for (int kc = 0; kc < D_MODEL / 32; ++kc) {
        __syncthreads();
        // ---- stage A chunk (128 x 32 bf16) ----
        if (AMODE == 0) {
            const float* A = (const float*)Aptr;
#pragma unroll
            for (int p = 0; p < 4; ++p) {
                int idx = p * 256 + tid;
                int r   = idx >> 3;
                int c4  = (idx & 7) * 4;
                float4 f = *(const float4*)(A + (size_t)(m0 + r) * D_MODEL + kc * 32 + c4);
                __bf16* d = &sA[r * 32 + c4];
                d[0] = (__bf16)f.x; d[1] = (__bf16)f.y;
                d[2] = (__bf16)f.z; d[3] = (__bf16)f.w;
            }
        } else {
            const __bf16* A = (const __bf16*)Aptr;
#if HAVE_TDM
            if (w == 0)   // one wave issues the async tensor DMA (EXEC ignored)
                tdm_load_tile_bf16(A + (size_t)m0 * D_MODEL + kc * 32,
                                   /*lds_off=*/0u, /*tile_w=*/32,
                                   /*tile_h=*/128, /*stride=*/D_MODEL);
#else
#pragma unroll
            for (int p = 0; p < 4; ++p) {
                int idx = p * 256 + tid;
                int r   = idx >> 3;
                int c4  = (idx & 7) * 4;
                *(uint2*)&sA[r * 32 + c4] =
                    *(const uint2*)(A + (size_t)(m0 + r) * D_MODEL + kc * 32 + c4);
            }
#endif
        }
        // ---- stage W chunk (32 x 64 f32) transposed into sB[n][k] ----
#pragma unroll
        for (int p = 0; p < 2; ++p) {
            int idx = p * 256 + tid;
            int kk  = idx >> 4;
            int c4  = (idx & 15) * 4;
            const float* wp = W + (size_t)(kc * 32 + kk) * D_MODEL + n0 + c4;
            float4 f = *(const float4*)wp;
            if (p == 0 && kc < D_MODEL / 32 - 1)
                __builtin_prefetch(wp + 32 * D_MODEL, 0, 1);  // global_prefetch_b8
            sB[(c4 + 0) * 32 + kk] = (__bf16)f.x;
            sB[(c4 + 1) * 32 + kk] = (__bf16)f.y;
            sB[(c4 + 2) * 32 + kk] = (__bf16)f.z;
            sB[(c4 + 3) * 32 + kk] = (__bf16)f.w;
        }
#if HAVE_TDM
        if (AMODE == 1 && w == 0)
            __builtin_amdgcn_s_wait_tensorcnt(0);  // TDM done before barrier
#endif
        __syncthreads();

        Frag a[2], b[2];
#pragma unroll
        for (int i = 0; i < 2; ++i) a[i] = load_frag(&sA[(wm * 32 + 16 * i) * 32], 32, lane);
#pragma unroll
        for (int j = 0; j < 2; ++j) b[j] = load_frag(&sB[(wn * 32 + 16 * j) * 32], 32, lane);
#pragma unroll
        for (int i = 0; i < 2; ++i)
#pragma unroll
            for (int j = 0; j < 2; ++j)
                acc[i][j] = wmma_bf16(a[i], b[j], acc[i][j]);
    }

    // ---- epilogue: C layout: col = lane%16, row = v + 8*(lane>=16) ----
    const int col = lane & 15;
    const int rb  = (lane & 16) ? 8 : 0;
#pragma unroll
    for (int i = 0; i < 2; ++i)
#pragma unroll
        for (int j = 0; j < 2; ++j) {
            int gn = n0 + wn * 32 + 16 * j + col;
            float bv = bias[gn];
#pragma unroll
            for (int v = 0; v < 8; ++v) {
                int gm = m0 + wm * 32 + 16 * i + v + rb;
                float val = acc[i][j][v] + bv;
                if (OMODE == 1) {
                    ((float*)Out)[(size_t)gm * D_MODEL + gn] = val;
                } else {
                    int b  = gm >> 11;
                    int s  = gm & (SEQ - 1);
                    int h  = gn >> 7;
                    int dh = gn & (D_HEAD - 1);
                    if (OMODE == 0)
                        ((__bf16*)Out)[(((size_t)(b * NUM_HEADS + h)) * SEQ + s) * D_HEAD + dh] = (__bf16)val;
                    else
                        ((__bf16*)Out)[(((size_t)(b * NUM_HEADS + h)) * D_HEAD + dh) * SEQ + s] = (__bf16)val;
                }
            }
        }
}

// ---------------------------------------------------------------------------
// Fused attention: 4 waves per block cooperate on 16 query rows of one (b,h).
//   Phase 1: scores -> 16 x 2048 f32 strip in LDS (128 KB); waves split S.
//   Phase 2: row softmax (4 rows/wave); attention written to d_out; bf16 P
//            packed IN PLACE into the f32 strip (write idx c trails read 2c).
//   Phase 3: context = P @ V (V pre-transposed); waves split the 8 Dh-tiles.
// LDS = 128 KB -> 2 blocks / WGP -> 8 resident waves.
// ---------------------------------------------------------------------------
__global__ __launch_bounds__(128)
void attention_fused(const __bf16* __restrict__ qws, const __bf16* __restrict__ kws,
                     const __bf16* __restrict__ vwsT, float* __restrict__ attn_out,
                     __bf16* __restrict__ ctx_ws) {
    extern __shared__ char smem[];
    float* sS = (float*)smem;   // 16 x 2048 f32; later rows reused as bf16 P

    const int tid  = threadIdx.x;
    const int lane = tid & 31;
    const int w    = tid >> 5;          // 0..3
    const int q0   = blockIdx.x * 16;
    const int bh   = blockIdx.y;

    const __bf16* qp = qws + (size_t)bh * SEQ * D_HEAD;
    const __bf16* kp = kws + (size_t)bh * SEQ * D_HEAD;
    const __bf16* vp = vwsT + (size_t)bh * D_HEAD * SEQ;
    const float scale = 0.08838834764831845f;   // 1/sqrt(128)

    Frag qa[4];
#pragma unroll
    for (int c = 0; c < 4; ++c)
        qa[c] = load_frag(qp + (size_t)q0 * D_HEAD + c * 32, D_HEAD, lane);

    // ---- Phase 1: scores; wave w owns n-tiles [32w, 32w+32) ----
    const int col = lane & 15;
    const int rb  = (lane & 16) ? 8 : 0;
    for (int nt = w * 32; nt < w * 32 + 32; ++nt) {
        v8f acc = {};
#pragma unroll
        for (int c = 0; c < 4; ++c) {
            Frag kb = load_frag(kp + (size_t)(nt * 16) * D_HEAD + c * 32, D_HEAD, lane);
            acc = wmma_bf16(qa[c], kb, acc);
        }
        int gc = nt * 16 + col;
#pragma unroll
        for (int v = 0; v < 8; ++v) sS[(v + rb) * SEQ + gc] = acc[v] * scale;
    }
    __syncthreads();

    // ---- Phase 2: softmax; wave w owns rows [4w, 4w+4) ----
    for (int r = w * 4; r < w * 4 + 4; ++r) {
        float mx = -1e30f;
        for (int c = lane; c < SEQ; c += 32) mx = fmaxf(mx, sS[r * SEQ + c]);
#pragma unroll
        for (int o = 16; o > 0; o >>= 1) mx = fmaxf(mx, __shfl_xor(mx, o, 32));
        float sum = 0.f;
        for (int c = lane; c < SEQ; c += 32) {
            float e = __expf(sS[r * SEQ + c] - mx);
            sS[r * SEQ + c] = e;
            sum += e;
        }
#pragma unroll
        for (int o = 16; o > 0; o >>= 1) sum += __shfl_xor(sum, o, 32);
        float inv = 1.f / sum;

        float*  arow = attn_out + ((size_t)bh * SEQ + q0 + r) * SEQ;
        __bf16* prow = (__bf16*)(sS + (size_t)r * SEQ);   // in-place bf16 row
        for (int c2 = lane; c2 < SEQ / 2; c2 += 32) {
            float p0 = sS[r * SEQ + 2 * c2] * inv;
            float p1 = sS[r * SEQ + 2 * c2 + 1] * inv;
            ((float2*)arow)[c2] = make_float2(p0, p1);
            PkBf pk; pk.h[0] = (__bf16)p0; pk.h[1] = (__bf16)p1;
            *(unsigned int*)&prow[2 * c2] = pk.u;         // c2 trails 2*c2: safe
        }
    }
    __syncthreads();

    // ---- Phase 3: context = P @ V; wave w owns Dh-tiles {2w, 2w+1} ----
    // bf16 P row r lives at byte offset r*8192 -> element stride 4096.
    const __bf16* pbase = (const __bf16*)sS;
    v8f acc2[2] = {};
    for (int kc = 0; kc < SEQ / 32; ++kc) {
        Frag pa = load_frag(pbase + kc * 32, 2 * SEQ, lane);
#pragma unroll
        for (int t = 0; t < 2; ++t) {
            int j = 2 * w + t;
            Frag vb = load_frag(vp + (size_t)(j * 16) * SEQ + kc * 32, SEQ, lane);
            acc2[t] = wmma_bf16(pa, vb, acc2[t]);
        }
    }
    const int b = bh >> 4, h = bh & 15;
#pragma unroll
    for (int t = 0; t < 2; ++t)
#pragma unroll
        for (int v = 0; v < 8; ++v) {
            int s  = q0 + v + rb;
            int dh = (2 * w + t) * 16 + col;
            ctx_ws[((size_t)b * SEQ + s) * D_MODEL + h * D_HEAD + dh] = (__bf16)acc2[t][v];
        }
}

// ---------------------------------------------------------------------------
extern "C" void kernel_launch(void* const* d_in, const int* in_sizes, int n_in,
                              void* d_out, int out_size, void* d_ws, size_t ws_size,
                              hipStream_t stream) {
    const float* query = (const float*)d_in[0];
    const float* key   = (const float*)d_in[1];
    const float* value = (const float*)d_in[2];
    const float* Wq = (const float*)d_in[3];
    const float* bq = (const float*)d_in[4];
    const float* Wk = (const float*)d_in[5];
    const float* bk = (const float*)d_in[6];
    const float* Wv = (const float*)d_in[7];
    const float* bv = (const float*)d_in[8];
    const float* Wo = (const float*)d_in[9];
    const float* bo = (const float*)d_in[10];

    const size_t Q = (size_t)BH * SEQ * D_HEAD;   // bf16 elements per buffer
    __bf16* qws = (__bf16*)d_ws;
    __bf16* kws = qws + Q;
    __bf16* vws = kws + Q;   // transposed (BH, Dh, S)
    __bf16* ctx = vws + Q;   // (B, S, D)

    float* out  = (float*)d_out;
    float* attn = out + (size_t)BATCH * SEQ * D_MODEL;

    dim3 gg(D_MODEL / 64, (BATCH * SEQ) / 128);
    dim3 gb(256);
    proj_gemm<0, 0><<<gg, gb, 0, stream>>>(query, Wq, bq, qws);
    proj_gemm<0, 0><<<gg, gb, 0, stream>>>(key,   Wk, bk, kws);
    proj_gemm<0, 2><<<gg, gb, 0, stream>>>(value, Wv, bv, vws);

    const size_t att_lds = (size_t)16 * SEQ * sizeof(float);   // 128 KB
    attention_fused<<<dim3(SEQ / 16, BH), 128, att_lds, stream>>>(qws, kws, vws, attn, ctx);

    proj_gemm<1, 1><<<gg, gb, 0, stream>>>(ctx, Wo, bo, out);
}
